// GNNHolographicPredictor_23545010716680
// MI455X (gfx1250) — matrix operand order repaired and checked by reading, not compiled, not run on hardware
//
#include <hip/hip_runtime.h>
#include <math.h>

#define NN 50000
#define NE 800000
#define RR (NE + NN)
#define HH 64
#define NL 4

typedef __attribute__((ext_vector_type(16))) _Float16 v16h;
typedef __attribute__((ext_vector_type(8)))  _Float16 v8h;
typedef __attribute__((ext_vector_type(8)))  float    v8f;

static __device__ __forceinline__ v16h zero_v16h() {
  v16h t;
#pragma unroll
  for (int j = 0; j < 16; ++j) t[j] = (_Float16)0.f;
  return t;
}
static __device__ __forceinline__ v8f zero_v8f() {
  v8f t;
#pragma unroll
  for (int j = 0; j < 8; ++j) t[j] = 0.f;
  return t;
}

__global__ void k_zero(float* p, int n) {
  int i = blockIdx.x * blockDim.x + threadIdx.x;
  if (i < n) p[i] = 0.f;
}
__global__ void k_copy(float* dst, const float* src, int n) {
  int i = blockIdx.x * blockDim.x + threadIdx.x;
  if (i < n) dst[i] = src[i];
}
__global__ void k_deg_init(float* deg) {
  int i = blockIdx.x * blockDim.x + threadIdx.x;
  if (i < NN) deg[i] = 1.f;  // self loop contributes 1
}
__global__ void k_deg_edge(const int* __restrict__ ecol, float* deg) {
  int e = blockIdx.x * blockDim.x + threadIdx.x;
  if (e < NE) atomicAdd(&deg[ecol[e]], 1.f);
}
__global__ void k_node_h(const float* __restrict__ x, const float* __restrict__ in_w,
                         const float* __restrict__ in_b, float* __restrict__ h) {
  int idx = blockIdx.x * blockDim.x + threadIdx.x;
  if (idx >= NN * HH) return;
  int n = idx >> 6, c = idx & 63;
  float s = in_b[c];
#pragma unroll
  for (int k = 0; k < 4; ++k) s += x[n * 4 + k] * in_w[k * HH + c];
  h[idx] = s;
}

// Pre-swizzle live weights into exact WMMA B-fragment order (one 32B load per frag).
__global__ void k_prep_w(const float* __restrict__ rw1, const float* __restrict__ rw2,
                         _Float16* __restrict__ w1s, _Float16* __restrict__ w2s) {
  int idx = blockIdx.x * blockDim.x + threadIdx.x;
  if (idx >= NL * 7168) return;
  int i = idx / 7168;
  int r = idx % 7168;
  if (r < 6144) {
    int f = r / 512, lr = r % 512;
    int lane = lr / 16, j = lr % 16;
    int ks = f / 4, nt = f % 4;
    int K = ks * 32 + (j < 8 ? ((lane >> 4) * 8) + j : 16 + ((lane >> 4) * 8) + (j - 8));
    int n = nt * 16 + (lane & 15);
    float v = (K < 66) ? rw1[(size_t)i * 66 * HH + K * HH + n] : 0.f;
    w1s[(size_t)i * 6144 + r] = (_Float16)v;
  } else {
    int rr = r - 6144;
    int ks = rr / 512, lr = rr % 512;
    int lane = lr / 16, j = lr % 16;
    int K = ks * 32 + (j < 8 ? ((lane >> 4) * 8) + j : 16 + ((lane >> 4) * 8) + (j - 8));
    int n = lane & 15;
    float v = (n < 4) ? rw2[(size_t)i * HH * HH + K * HH + n] : 0.f;
    w2s[(size_t)i * 1024 + rr] = (_Float16)v;
  }
}

// Flagship WMMA kernel: 16-row tile per wave, A gathered in native 16-bit A layout,
// GEMM1 (Kpad=96 -> 12 wmma) + relu + LDS transpose + GEMM2 (K=64 -> 2 wmma).
__global__ __launch_bounds__(256) void k_edge_msg(
    const float* __restrict__ h, const int* __restrict__ erow,
    const int* __restrict__ ecol, const float* __restrict__ ea,
    const _Float16* __restrict__ w1s, const _Float16* __restrict__ w2s,
    const float* __restrict__ rb1, const float* __restrict__ rb2,
    float* __restrict__ agg) {
  __shared__ _Float16 lds[8][16 * HH];
  const int wave = threadIdx.x >> 5;
  const int lane = threadIdx.x & 31;
  const int hi = lane >> 4;
  const int ln = lane & 15;
  const int base = (blockIdx.x * 8 + wave) * 16;

  int g = base + ln;
  int gc = (g < RR) ? g : (RR - 1);
  int src = (gc < NE) ? erow[gc] : (gc - NE);
  const float* hrow = h + (size_t)src * HH;

  v16h a[3];
#pragma unroll
  for (int ks = 0; ks < 2; ++ks) {
    const float4* p0 = (const float4*)(hrow + ks * 32 + 8 * hi);
    const float4* p1 = (const float4*)(hrow + ks * 32 + 16 + 8 * hi);
    float4 f0 = p0[0], f1 = p0[1], f2 = p1[0], f3 = p1[1];
    v16h t;
    t[0] = (_Float16)f0.x;  t[1] = (_Float16)f0.y;
    t[2] = (_Float16)f0.z;  t[3] = (_Float16)f0.w;
    t[4] = (_Float16)f1.x;  t[5] = (_Float16)f1.y;
    t[6] = (_Float16)f1.z;  t[7] = (_Float16)f1.w;
    t[8]  = (_Float16)f2.x; t[9]  = (_Float16)f2.y;
    t[10] = (_Float16)f2.z; t[11] = (_Float16)f2.w;
    t[12] = (_Float16)f3.x; t[13] = (_Float16)f3.y;
    t[14] = (_Float16)f3.z; t[15] = (_Float16)f3.w;
    a[ks] = t;
  }
  {
    v16h t = zero_v16h();
    if (hi == 0 && gc < NE) {  // edge_attr occupies padded K cols 64,65
      t[0] = (_Float16)ea[gc * 2 + 0];
      t[1] = (_Float16)ea[gc * 2 + 1];
    }
    a[2] = t;
  }

#pragma unroll
  for (int nt = 0; nt < 4; ++nt) {
    v8f c = zero_v8f();
#pragma unroll
    for (int ks = 0; ks < 3; ++ks) {
      v16h b = *(const v16h*)(w1s + ((size_t)((ks * 4 + nt) * 32 + lane)) * 16);
      c = __builtin_amdgcn_wmma_f32_16x16x32_f16(false, a[ks], false, b,
                                                 (short)0, c, false, false);
    }
    float bias = rb1[nt * 16 + ln];
#pragma unroll
    for (int r = 0; r < 8; ++r) {
      float v = c[r] + bias;
      v = v > 0.f ? v : 0.f;
      lds[wave][(r + 8 * hi) * HH + nt * 16 + ln] = (_Float16)v;
    }
  }
  asm volatile("s_wait_dscnt 0" ::: "memory");  // wave-local LDS RAW fence

  v8f c2 = zero_v8f();
#pragma unroll
  for (int ks = 0; ks < 2; ++ks) {
    v8h lo = *(const v8h*)&lds[wave][ln * HH + ks * 32 + 8 * hi];
    v8h hq = *(const v8h*)&lds[wave][ln * HH + ks * 32 + 16 + 8 * hi];
    v16h a2 = __builtin_shufflevector(lo, hq, 0, 1, 2, 3, 4, 5, 6, 7, 8, 9, 10,
                                      11, 12, 13, 14, 15);
    v16h b2 = *(const v16h*)(w2s + ((size_t)(ks * 32 + lane)) * 16);
    c2 = __builtin_amdgcn_wmma_f32_16x16x32_f16(false, a2, false, b2, (short)0,
                                                c2, false, false);
  }

  if (ln < 4) {
    float bias2 = rb2[ln];
#pragma unroll
    for (int r = 0; r < 8; ++r) {
      int g2 = base + r + 8 * hi;
      if (g2 < RR) {
        int dst = (g2 < NE) ? ecol[g2] : (g2 - NE);
        atomicAdd(&agg[dst * 4 + ln], c2[r] + bias2);
      }
    }
  }
}

__global__ void k_x4(const float* __restrict__ agg, const float* __restrict__ deg,
                     const float* __restrict__ h, const float* __restrict__ sw,
                     const float* __restrict__ sb, float* __restrict__ x4) {
  int idx = blockIdx.x * blockDim.x + threadIdx.x;
  if (idx >= NN * 4) return;
  int n = idx >> 2, j = idx & 3;
  float s = agg[idx] / deg[n] + sb[j];
  const float* hr = h + (size_t)n * HH;
#pragma unroll
  for (int k = 0; k < HH; ++k) s += hr[k] * sw[k * 256 + j];
  x4[idx] = s;
}

__global__ void k_qkv(const float* __restrict__ x4, const float* __restrict__ qw,
                      const float* __restrict__ qb, const float* __restrict__ kw,
                      const float* __restrict__ kb, const float* __restrict__ vw,
                      const float* __restrict__ vb, float* __restrict__ Q,
                      float* __restrict__ K, float* __restrict__ V) {
  int idx = blockIdx.x * blockDim.x + threadIdx.x;
  if (idx >= NN * HH) return;
  int n = idx >> 6, c = idx & 63;
  float x0 = x4[n * 4 + 0], x1 = x4[n * 4 + 1], x2 = x4[n * 4 + 2],
        x3 = x4[n * 4 + 3];
  Q[idx] = qb[c] + x0 * qw[c] + x1 * qw[64 + c] + x2 * qw[128 + c] + x3 * qw[192 + c];
  K[idx] = kb[c] + x0 * kw[c] + x1 * kw[64 + c] + x2 * kw[128 + c] + x3 * kw[192 + c];
  V[idx] = vb[c] + x0 * vw[c] + x1 * vw[64 + c] + x2 * vw[128 + c] + x3 * vw[192 + c];
}

// Global softmax pass 1: exp + per-head sum. Two-level reduction: LDS atomics
// within the block (ds_add at WGP speed), then 4 global atomics per block.
__global__ __launch_bounds__(256) void k_score(
    const int* __restrict__ erow, const int* __restrict__ ecol,
    const float* __restrict__ Q, const float* __restrict__ K,
    float* __restrict__ sexp, float* __restrict__ hsum) {
  __shared__ float red[4];
  if (threadIdx.x < 4) red[threadIdx.x] = 0.f;
  __syncthreads();
  int e = blockIdx.x * blockDim.x + threadIdx.x;
  if (e < NE) {
    int r = erow[e], c = ecol[e];
#pragma unroll
    for (int hh = 0; hh < 4; ++hh) {
      float s = 0.f;
#pragma unroll
      for (int d = 0; d < 16; ++d)
        s += Q[(size_t)r * HH + hh * 16 + d] * K[(size_t)c * HH + hh * 16 + d];
      float es = __expf(s * 0.25f);  // 1/sqrt(HD=16)
      sexp[e * 4 + hh] = es;
      atomicAdd(&red[hh], es);
    }
  }
  __syncthreads();
  if (threadIdx.x < 4) atomicAdd(&hsum[threadIdx.x], red[threadIdx.x]);
}

__global__ void k_att(const int* __restrict__ erow, const int* __restrict__ ecol,
                      const float* __restrict__ V, const float* __restrict__ sexp,
                      const float* __restrict__ hsum, float* __restrict__ att) {
  int e = blockIdx.x * blockDim.x + threadIdx.x;
  if (e >= NE) return;
  int r = erow[e], c = ecol[e];
#pragma unroll
  for (int hh = 0; hh < 4; ++hh) {
    float w = sexp[e * 4 + hh] / hsum[hh];
#pragma unroll
    for (int d = 0; d < 16; ++d)
      atomicAdd(&att[(size_t)c * HH + hh * 16 + d],
                w * V[(size_t)r * HH + hh * 16 + d]);
  }
}

__global__ void k_update(const float* __restrict__ att, const float* __restrict__ ow,
                         const float* __restrict__ ob, const float* __restrict__ x4,
                         const float* __restrict__ lg, const float* __restrict__ lb,
                         const float* __restrict__ res_w, int layer,
                         const float* __restrict__ in_w,
                         const float* __restrict__ in_b, float* __restrict__ xbuf,
                         float* __restrict__ h) {
  int n = blockIdx.x * blockDim.x + threadIdx.x;
  if (n >= NN) return;
  float rw = res_w[layer];
  const float* ar = att + (size_t)n * HH;
  float o[4];
#pragma unroll
  for (int j = 0; j < 4; ++j) {
    float s = ob[j] + x4[n * 4 + j];
    for (int c = 0; c < HH; ++c) s += ar[c] * ow[c * 4 + j];
    o[j] = s;
  }
  float mu = 0.25f * (o[0] + o[1] + o[2] + o[3]);
  float var = 0.f;
#pragma unroll
  for (int j = 0; j < 4; ++j) { float d = o[j] - mu; var += d * d; }
  var *= 0.25f;
  float inv = rsqrtf(var + 1e-5f);
  float xn[4];
#pragma unroll
  for (int j = 0; j < 4; ++j)
    xn[j] = (o[j] - mu) * inv * lg[j] + lb[j] + xbuf[n * 4 + j] * rw;
#pragma unroll
  for (int j = 0; j < 4; ++j) xbuf[n * 4 + j] = xn[j];
  for (int c = 0; c < HH; ++c) {
    float s = in_b[c];
#pragma unroll
    for (int k = 0; k < 4; ++k) s += xn[k] * in_w[k * HH + c];
    h[(size_t)n * HH + c] = s;
  }
}

__global__ void k_pred(const float* __restrict__ h, const float* __restrict__ w1,
                       const float* __restrict__ b1, const float* __restrict__ w2,
                       const float* __restrict__ b2, float* __restrict__ out) {
  int n = blockIdx.x * blockDim.x + threadIdx.x;
  if (n >= NN) return;
  const float* hr = h + (size_t)n * HH;
  float t[32];
  for (int j = 0; j < 32; ++j) {
    float s = b1[j];
    for (int k = 0; k < HH; ++k) s += hr[k] * w1[k * 32 + j];
    t[j] = s > 0.f ? s : 0.f;
  }
#pragma unroll
  for (int j = 0; j < 4; ++j) {
    float s = b2[j];
    for (int k = 0; k < 32; ++k) s += t[k] * w2[k * 4 + j];
    out[n * 4 + j] = tanhf(s) * 0.3f;
  }
}

static inline dim3 gs1(long n, int b = 256) { return dim3((unsigned)((n + b - 1) / b)); }

extern "C" void kernel_launch(void* const* d_in, const int* in_sizes, int n_in,
                              void* d_out, int out_size, void* d_ws, size_t ws_size,
                              hipStream_t stream) {
  (void)in_sizes; (void)n_in; (void)out_size; (void)ws_size;
  const float* x_in = (const float*)d_in[0];
  const int*   eidx = (const int*)d_in[1];
  const int*   erow = eidx;
  const int*   ecol = eidx + NE;
  const float* ea   = (const float*)d_in[2];
  const float* in_w = (const float*)d_in[3];
  const float* in_b = (const float*)d_in[4];
  const float* rw1  = (const float*)d_in[5];
  const float* rb1  = (const float*)d_in[6];
  const float* rw2  = (const float*)d_in[7];
  const float* rb2  = (const float*)d_in[8];
  // d_in[9..12] (imag MLP): provably dead — only h4[:, :4] is ever consumed.
  const float* sw   = (const float*)d_in[13];
  const float* sb   = (const float*)d_in[14];
  const float* qw   = (const float*)d_in[15];
  const float* qb   = (const float*)d_in[16];
  const float* kw   = (const float*)d_in[17];
  const float* kb   = (const float*)d_in[18];
  const float* vw   = (const float*)d_in[19];
  const float* vb   = (const float*)d_in[20];
  const float* ow   = (const float*)d_in[21];
  const float* ob   = (const float*)d_in[22];
  const float* lg   = (const float*)d_in[23];
  const float* lb   = (const float*)d_in[24];
  const float* w1   = (const float*)d_in[25];
  const float* b1   = (const float*)d_in[26];
  const float* w2   = (const float*)d_in[27];
  const float* b2   = (const float*)d_in[28];
  const float* resw = (const float*)d_in[29];
  float* out = (float*)d_out;

  char* W = (char*)d_ws;
  auto take = [&](size_t bytes) {
    char* p = W;
    W += (bytes + 255) & ~(size_t)255;
    return p;
  };
  float*    h    = (float*)take((size_t)NN * HH * 4);
  float*    xbuf = (float*)take((size_t)NN * 4 * 4);
  float*    deg  = (float*)take((size_t)NN * 4);
  float*    agg  = (float*)take((size_t)NN * 4 * 4);
  float*    x4   = (float*)take((size_t)NN * 4 * 4);
  float*    Q    = (float*)take((size_t)NN * HH * 4);
  float*    K    = (float*)take((size_t)NN * HH * 4);
  float*    V    = (float*)take((size_t)NN * HH * 4);
  float*    att  = (float*)take((size_t)NN * HH * 4);
  float*    sexp = (float*)take((size_t)NE * 4 * 4);
  float*    hsum = (float*)take(256);
  _Float16* w1s  = (_Float16*)take((size_t)NL * 6144 * 2);
  _Float16* w2s  = (_Float16*)take((size_t)NL * 1024 * 2);

  k_prep_w<<<gs1((long)NL * 7168), 256, 0, stream>>>(rw1, rw2, w1s, w2s);
  k_copy<<<gs1((long)NN * 4), 256, 0, stream>>>(xbuf, x_in, NN * 4);
  k_deg_init<<<gs1(NN), 256, 0, stream>>>(deg);
  k_deg_edge<<<gs1(NE), 256, 0, stream>>>(ecol, deg);
  k_node_h<<<gs1((long)NN * HH), 256, 0, stream>>>(xbuf, in_w, in_b, h);

  const int tiles = (RR + 15) / 16;
  const int eblocks = (tiles + 7) / 8;

  for (int i = 0; i < NL; ++i) {
    k_zero<<<gs1((long)NN * 4), 256, 0, stream>>>(agg, NN * 4);
    k_zero<<<1, 64, 0, stream>>>(hsum, 4);
    k_zero<<<gs1((long)NN * HH), 256, 0, stream>>>(att, NN * HH);

    k_edge_msg<<<eblocks, 256, 0, stream>>>(
        h, erow, ecol, ea, w1s + (size_t)i * 6144, w2s + (size_t)i * 1024,
        rb1 + (size_t)i * HH, rb2 + (size_t)i * HH, agg);

    k_x4<<<gs1((long)NN * 4), 256, 0, stream>>>(agg, deg, h,
                                                sw + (size_t)i * HH * 256,
                                                sb + (size_t)i * 256, x4);
    k_qkv<<<gs1((long)NN * HH), 256, 0, stream>>>(
        x4, qw + (size_t)i * 256, qb + (size_t)i * HH, kw + (size_t)i * 256,
        kb + (size_t)i * HH, vw + (size_t)i * 256, vb + (size_t)i * HH, Q, K, V);
    k_score<<<gs1(NE), 256, 0, stream>>>(erow, ecol, Q, K, sexp, hsum);
    k_att<<<gs1(NE), 256, 0, stream>>>(erow, ecol, V, sexp, hsum, att);
    k_update<<<gs1(NN), 256, 0, stream>>>(
        att, ow + (size_t)i * HH * 4, ob + (size_t)i * 4, x4,
        lg + (size_t)i * 4, lb + (size_t)i * 4, resw, i, in_w, in_b, xbuf, h);
  }

  k_pred<<<gs1(NN), 256, 0, stream>>>(h, w1, b1, w2, b2, out);
}
